// SelectiveStateSpace_10763188044007
// MI455X (gfx1250) — matrix operand order repaired and checked by reading, not compiled, not run on hardware
//
#include <hip/hip_runtime.h>

// ---------------- types ----------------
typedef __bf16 bf16;
typedef __attribute__((ext_vector_type(16))) __bf16 v16bf;
typedef __attribute__((ext_vector_type(8)))  float   v8f;
typedef __attribute__((ext_vector_type(4)))  float   v4f;
typedef __attribute__((ext_vector_type(4)))  unsigned int v4u;
typedef __attribute__((ext_vector_type(4)))  int v4i;

// address-space-qualified int4 pointers for the async global->LDS builtin
typedef __attribute__((address_space(1))) v4i* gas_v4i;
typedef __attribute__((address_space(3))) v4i* las_v4i;

union Frag { v16bf bf; v4u u[2]; };

// ---------------- problem constants ----------------
#define BATCH 2
#define TLEN  2048
#define DDIM  1024
#define NSTATE 16
#define RRANK 64
#define XDBL  96           // R + 2N
#define BT    (BATCH*TLEN) // 4096
#define TWOD  (2*DDIM)     // 2048
#define CHUNK 128
#define NCHUNK (TLEN/CHUNK) // 16

// CDNA5 async global->LDS path (probe-confirmed on this toolchain)
#if defined(__has_builtin)
#if __has_builtin(__builtin_amdgcn_global_load_async_to_lds_b128)
#define HAVE_ASYNC_LDS 1
#endif
#endif

__device__ __forceinline__ void wait_asynccnt0() {
#if defined(__has_builtin)
#if __has_builtin(__builtin_amdgcn_s_wait_asynccnt)
    __builtin_amdgcn_s_wait_asynccnt(0);
    return;
#else
    asm volatile("s_wait_asynccnt 0x0" ::: "memory");
    return;
#endif
#else
    asm volatile("s_wait_asynccnt 0x0" ::: "memory");
#endif
}

__device__ __forceinline__ void sched_fence() {
#if defined(__has_builtin)
#if __has_builtin(__builtin_amdgcn_sched_barrier)
    __builtin_amdgcn_sched_barrier(0);
    return;
#else
    asm volatile("" ::: "memory");
    return;
#endif
#else
    asm volatile("" ::: "memory");
#endif
}

// ---------------- helpers ----------------
__device__ __forceinline__ float siluf(float x) {
    return x / (1.0f + __expf(-x));
}
__device__ __forceinline__ float softplusf(float x) {
    if (x > 20.0f) return x;
    return log1pf(__expf(x));
}

// ---------------- f32 -> bf16 convert ----------------
__global__ void cvt_bf16_kernel(const float* __restrict__ in, bf16* __restrict__ out, int n) {
    int i = blockIdx.x * blockDim.x + threadIdx.x;
    if (i < n) out[i] = (bf16)in[i];
}

// ---------------- WMMA GEMM: C[m,n] = sum_k A[m,k]*B[n,k]  (TN, row-major) ---
// A: M x K bf16, B: N x K bf16, Cout: M x N f32.
// epi: 0 = none; 1 = softplus(acc + aux[col]) + 1e-5; 2 = acc * silu(aux[row*aux_ld+col])
// Block = 256 threads (8 waves). Block tile 128x128, K-step 32, double-buffered LDS,
// async global->LDS fetch (ASYNCcnt), single compute site, single-lookahead pipeline.
#define LDS_STRIDE 40   // 32 + 8 pad (bf16 units) -> conflict-free row access
__global__ __launch_bounds__(256)
void wmma_gemm_tn(const bf16* __restrict__ A, const bf16* __restrict__ Bm,
                  float* __restrict__ Cout, int M, int N, int K,
                  int epi, const float* __restrict__ aux, int aux_ld) {
    __shared__ bf16 As[2][128 * LDS_STRIDE];
    __shared__ bf16 Bs[2][128 * LDS_STRIDE];

    const int t    = threadIdx.x;
    const int lane = t & 31;
    const int wave = t >> 5;
    const int r    = lane & 15;
    const int hi   = lane >> 4;

    const int m0 = blockIdx.y * 128;
    const int n0 = blockIdx.x * 128;
    const int nk = K >> 5;
    const bool fullN = (n0 + 128 <= N);   // uniform: skip per-row guards in hot loop

    const v8f vzero = {0.f,0.f,0.f,0.f,0.f,0.f,0.f,0.f};
    v8f acc[8];
#pragma unroll
    for (int i = 0; i < 8; ++i) acc[i] = vzero;

    const v4u z4 = {0u,0u,0u,0u};

    // Each thread moves 2 chunks (8 bf16 each) of A and of B per tile:
    // chunk c in [0,512): row = c>>2, col8 = (c&3)*8 ; this thread: c = t and c = t+256.
    const int srow = t >> 2;
    const int scol = (t & 3) * 8;

#if HAVE_ASYNC_LDS
    // ---- async fetch: global -> LDS directly, tracked by ASYNCcnt ----
    auto fetch = [&](int kt, int buf) {
        const int k0 = kt * 32;
        const bf16* Ap  = A + (size_t)(m0 + srow) * K + k0 + scol;
        bf16* la0 = &As[buf][srow * LDS_STRIDE + scol];
        bf16* la1 = &As[buf][(srow + 64) * LDS_STRIDE + scol];
        __builtin_amdgcn_global_load_async_to_lds_b128(
            (gas_v4i)Ap, (las_v4i)la0, 0, 0);
        __builtin_amdgcn_global_load_async_to_lds_b128(
            (gas_v4i)(Ap + (size_t)64 * K), (las_v4i)la1, 0, 0);
        const bf16* Bp = Bm + (size_t)(n0 + srow) * K + k0 + scol;
        bf16* lb0 = &Bs[buf][srow * LDS_STRIDE + scol];
        bf16* lb1 = &Bs[buf][(srow + 64) * LDS_STRIDE + scol];
        if (fullN) {           // uniform fast path: no exec-mask churn
            __builtin_amdgcn_global_load_async_to_lds_b128(
                (gas_v4i)Bp, (las_v4i)lb0, 0, 0);
            __builtin_amdgcn_global_load_async_to_lds_b128(
                (gas_v4i)(Bp + (size_t)64 * K), (las_v4i)lb1, 0, 0);
        } else {
            const int nr0 = n0 + srow, nr1 = n0 + srow + 64;
            if (nr0 < N) {
                __builtin_amdgcn_global_load_async_to_lds_b128(
                    (gas_v4i)Bp, (las_v4i)lb0, 0, 0);
            } else {
                *(v4u*)lb0 = z4;
            }
            if (nr1 < N) {
                __builtin_amdgcn_global_load_async_to_lds_b128(
                    (gas_v4i)(Bp + (size_t)64 * K), (las_v4i)lb1, 0, 0);
            } else {
                *(v4u*)lb1 = z4;
            }
        }
    };
#else
    v4u sa[2], sb[2];   // one staging set (single-lookahead)
    auto gload = [&](int kt) {
        const int k0 = kt * 32;
        const bf16* Ap = A + (size_t)(m0 + srow) * K + k0 + scol;
        sa[0] = *(const v4u*)Ap;
        sa[1] = *(const v4u*)(Ap + (size_t)64 * K);
        const int nr0 = n0 + srow, nr1 = n0 + srow + 64;
        sb[0] = z4; sb[1] = z4;
        if (nr0 < N) sb[0] = *(const v4u*)(Bm + (size_t)nr0 * K + k0 + scol);
        if (nr1 < N) sb[1] = *(const v4u*)(Bm + (size_t)nr1 * K + k0 + scol);
    };
    auto sstore = [&](int buf) {
        *(v4u*)(&As[buf][srow * LDS_STRIDE + scol])        = sa[0];
        *(v4u*)(&As[buf][(srow + 64) * LDS_STRIDE + scol]) = sa[1];
        *(v4u*)(&Bs[buf][srow * LDS_STRIDE + scol])        = sb[0];
        *(v4u*)(&Bs[buf][(srow + 64) * LDS_STRIDE + scol]) = sb[1];
    };
#endif

    // ---- prologue: tile 0 into buffer 0 ----
#if HAVE_ASYNC_LDS
    fetch(0, 0);
    wait_asynccnt0();
#else
    gload(0);
    sstore(0);
#endif
    __syncthreads();

    // ---- main loop: ONE compute site so acc stays pinned ----
    for (int kt = 0; kt < nk; ++kt) {
        const int cur = kt & 1;
#if HAVE_ASYNC_LDS
        if (kt + 1 < nk) fetch(kt + 1, cur ^ 1);     // async, lands during compute
#else
        if (kt + 1 < nk) gload(kt + 1);              // vmem issued before compute
#endif
        // A fragment: 16x32; lane holds row (wave*16+r), K chunks at hi*8 and hi*8+16
        Frag af;
        const bf16* ap = &As[cur][(wave * 16 + r) * LDS_STRIDE + hi * 8];
        af.u[0] = *(const v4u*)ap;
        af.u[1] = *(const v4u*)(ap + 16);
        // Load ALL 8 B fragments...
        Frag bfr[8];
#pragma unroll
        for (int nt = 0; nt < 8; ++nt) {
            const bf16* bp = &Bs[cur][(nt * 16 + r) * LDS_STRIDE + hi * 8];
            bfr[nt].u[0] = *(const v4u*)bp;
            bfr[nt].u[1] = *(const v4u*)(bp + 16);
        }
        // ...then force: one dscnt wait, 8 WMMAs back-to-back on the matrix pipe.
        sched_fence();
#pragma unroll
        for (int nt = 0; nt < 8; ++nt) {
            acc[nt] = __builtin_amdgcn_wmma_f32_16x16x32_bf16(
                false, af.bf, false, bfr[nt].bf, (short)0, acc[nt], false, false);
        }
#if HAVE_ASYNC_LDS
        if (kt + 1 < nk) wait_asynccnt0();           // next tile resident in LDS
#else
        if (kt + 1 < nk) sstore(cur ^ 1);            // waits loadcnt, ds_store
#endif
        __syncthreads();
    }

    // ---- store with epilogue; lane layout: col = r, row = v + 8*hi ----
#pragma unroll
    for (int nt = 0; nt < 8; ++nt) {
        int col = n0 + nt * 16 + r;
        if (col >= N) continue;
#pragma unroll
        for (int v = 0; v < 8; ++v) {
            size_t row = (size_t)(m0 + wave * 16 + hi * 8 + v);
            float val = acc[nt][v];
            if (epi == 1) {
                val = softplusf(val + aux[col]) + 1e-5f;
            } else if (epi == 2) {
                val *= siluf(aux[row * (size_t)aux_ld + col]);
            }
            Cout[row * (size_t)N + col] = val;
        }
    }
}

// ---------------- depthwise conv(3) + SiLU ----------------
__global__ void conv_silu_kernel(const float* __restrict__ xz, const float* __restrict__ conv_w,
                                 float* __restrict__ xs_f, bf16* __restrict__ xs_b) {
    int idx = blockIdx.x * blockDim.x + threadIdx.x;     // over B*T*D
    if (idx >= BT * DDIM) return;
    int d  = idx & (DDIM - 1);
    int bt = idx >> 10;
    int t  = bt & (TLEN - 1);
    int b  = bt >> 11;
    const float* base = xz + (size_t)b * TLEN * TWOD;
    float xm = (t > 0)        ? base[(size_t)(t - 1) * TWOD + d] : 0.0f;
    float x0 =                  base[(size_t)t       * TWOD + d];
    float xp = (t < TLEN - 1) ? base[(size_t)(t + 1) * TWOD + d] : 0.0f;
    float w0 = conv_w[d * 3 + 0], w1 = conv_w[d * 3 + 1], w2 = conv_w[d * 3 + 2];
    float v = fmaf(xm, w0, fmaf(x0, w1, xp * w2));
    float s = siluf(v);
    xs_f[idx] = s;
    xs_b[idx] = (bf16)s;
}

// ---------------- dt_param slice -> bf16 ----------------
__global__ void dtp_kernel(const float* __restrict__ x_dbl, bf16* __restrict__ dtp_b) {
    int idx = blockIdx.x * blockDim.x + threadIdx.x;     // over BT*R
    if (idx >= BT * RRANK) return;
    int rr = idx & (RRANK - 1);
    int bt = idx >> 6;
    dtp_b[idx] = (bf16)x_dbl[(size_t)bt * XDBL + rr];
}

// ---------------- C = Cp @ W_C^T (K = 16, plain VALU) ----------------
__global__ void cmat_kernel(const float* __restrict__ x_dbl, const float* __restrict__ W_C,
                            float* __restrict__ cmat) {
    int idx = blockIdx.x * blockDim.x + threadIdx.x;     // over BT*D
    if (idx >= BT * DDIM) return;
    int d  = idx & (DDIM - 1);
    int bt = idx >> 10;
    const float* cp = x_dbl + (size_t)bt * XDBL + RRANK + NSTATE;  // cols 80..95
    const float* wc = W_C + (size_t)d * NSTATE;
    float s = 0.0f;
#pragma unroll
    for (int n = 0; n < NSTATE; ++n) s = fmaf(cp[n], wc[n], s);
    cmat[idx] = s;
}

// =====================================================================
// Chunked parallel scan (3 passes). thread id layout: id = (b*NCHUNK + c)*D + d
// =====================================================================
__global__ void scan_chunk_kernel(const float* __restrict__ x_dbl, const float* __restrict__ dt,
                                  const float* __restrict__ xs, const float* __restrict__ A_log,
                                  float* __restrict__ Pbuf, float* __restrict__ Qbuf) {
    int id = blockIdx.x * blockDim.x + threadIdx.x;      // 0 .. B*NCHUNK*D-1
    if (id >= BATCH * NCHUNK * DDIM) return;
    int d = id & (DDIM - 1);
    int c = (id >> 10) & (NCHUNK - 1);
    int b = id >> 14;

    float e[NSTATE], P[NSTATE], q[NSTATE];
#pragma unroll
    for (int n = 0; n < NSTATE; ++n) {
        e[n] = __expf(A_log[(size_t)d * NSTATE + n]);
        P[n] = 1.0f; q[n] = 0.0f;
    }
    const int t0 = c * CHUNK;
    const float* xd = x_dbl + (size_t)b * TLEN * XDBL;
    size_t base = (size_t)(b * TLEN + t0) * DDIM + d;

    for (int tt = 0; tt < CHUNK; ++tt) {
        size_t off = base + (size_t)tt * DDIM;
        float dtv = dt[off];
        float u   = dtv * xs[off];
        const v4f* bp = (const v4f*)(xd + (size_t)(t0 + tt) * XDBL + RRANK);
        v4f bpv[4] = { bp[0], bp[1], bp[2], bp[3] };
#pragma unroll
        for (int n = 0; n < NSTATE; ++n) {
            float a = __expf(-e[n] * dtv);
            P[n] *= a;
            q[n] = fmaf(a, q[n], bpv[n >> 2][n & 3] * u);
        }
    }
    v4f* Pd = (v4f*)(Pbuf + (size_t)id * NSTATE);
    v4f* Qd = (v4f*)(Qbuf + (size_t)id * NSTATE);
#pragma unroll
    for (int i = 0; i < 4; ++i) {
        v4f pv = { P[4*i], P[4*i+1], P[4*i+2], P[4*i+3] };
        v4f qv = { q[4*i], q[4*i+1], q[4*i+2], q[4*i+3] };
        Pd[i] = pv; Qd[i] = qv;
    }
}

__global__ void scan_combine_kernel(const float* __restrict__ Pbuf, const float* __restrict__ Qbuf,
                                    float* __restrict__ Sinit) {
    int id = blockIdx.x * blockDim.x + threadIdx.x;      // 0 .. B*D-1
    if (id >= BATCH * DDIM) return;
    int d = id & (DDIM - 1);
    int b = id >> 10;
    float S[NSTATE];
#pragma unroll
    for (int n = 0; n < NSTATE; ++n) S[n] = 0.0f;
    for (int c = 0; c < NCHUNK; ++c) {
        size_t idx = ((size_t)(b * NCHUNK + c) * DDIM + d) * NSTATE;
#pragma unroll
        for (int n = 0; n < NSTATE; ++n) {
            Sinit[idx + n] = S[n];
            S[n] = fmaf(Pbuf[idx + n], S[n], Qbuf[idx + n]);
        }
    }
}

__global__ void scan_emit_kernel(const float* __restrict__ x_dbl, const float* __restrict__ dt,
                                 const float* __restrict__ xs, const float* __restrict__ cmat,
                                 const float* __restrict__ A_log, const float* __restrict__ D_param,
                                 const float* __restrict__ Sinit, bf16* __restrict__ y_b) {
    int id = blockIdx.x * blockDim.x + threadIdx.x;      // 0 .. B*NCHUNK*D-1
    if (id >= BATCH * NCHUNK * DDIM) return;
    int d = id & (DDIM - 1);
    int c = (id >> 10) & (NCHUNK - 1);
    int b = id >> 14;

    float e[NSTATE], s[NSTATE];
    const float* Si = Sinit + (size_t)id * NSTATE;
#pragma unroll
    for (int n = 0; n < NSTATE; ++n) {
        e[n] = __expf(A_log[(size_t)d * NSTATE + n]);
        s[n] = Si[n];
    }
    float Dp = D_param[d];

    const int t0 = c * CHUNK;
    const float* xd = x_dbl + (size_t)b * TLEN * XDBL;
    size_t base = (size_t)(b * TLEN + t0) * DDIM + d;

    for (int tt = 0; tt < CHUNK; ++tt) {
        size_t off = base + (size_t)tt * DDIM;
        float dtv = dt[off];
        float xsv = xs[off];
        float cm  = cmat[off];
        const v4f* bp = (const v4f*)(xd + (size_t)(t0 + tt) * XDBL + RRANK);
        v4f bpv[4] = { bp[0], bp[1], bp[2], bp[3] };
        float u = dtv * xsv;
        float ssum = 0.0f;
#pragma unroll
        for (int n = 0; n < NSTATE; ++n) {
            float a = __expf(-e[n] * dtv);
            s[n] = fmaf(a, s[n], bpv[n >> 2][n & 3] * u);
            ssum += s[n];
        }
        y_b[off] = (bf16)(fmaf(cm, ssum, Dp * xsv));
    }
}

// ---------------- host launch ----------------
extern "C" void kernel_launch(void* const* d_in, const int* in_sizes, int n_in,
                              void* d_out, int out_size, void* d_ws, size_t ws_size,
                              hipStream_t stream) {
    (void)in_sizes; (void)n_in; (void)out_size; (void)ws_size;

    const float* x       = (const float*)d_in[0];
    const float* W_in    = (const float*)d_in[1];
    const float* conv_w  = (const float*)d_in[2];
    const float* W_x     = (const float*)d_in[3];
    const float* W_dt    = (const float*)d_in[4];
    const float* b_dt    = (const float*)d_in[5];
    const float* A_log   = (const float*)d_in[6];
    const float* D_param = (const float*)d_in[7];
    const float* W_C     = (const float*)d_in[8];
    const float* W_out   = (const float*)d_in[9];
    float* out = (float*)d_out;

    char* ws = (char*)d_ws;
    size_t off = 0;
    auto alloc = [&](size_t bytes) -> char* {
        off = (off + 255) & ~(size_t)255;
        char* p = ws + off;
        off += bytes;
        return p;
    };

    bf16*  xb      = (bf16*) alloc((size_t)BT * DDIM * 2);
    bf16*  W_in_b  = (bf16*) alloc((size_t)TWOD * DDIM * 2);
    bf16*  W_x_b   = (bf16*) alloc((size_t)XDBL * DDIM * 2);
    bf16*  W_dt_b  = (bf16*) alloc((size_t)DDIM * RRANK * 2);
    bf16*  W_out_b = (bf16*) alloc((size_t)DDIM * DDIM * 2);
    float* xz      = (float*)alloc((size_t)BT * TWOD * 4);
    float* xs_f    = (float*)alloc((size_t)BT * DDIM * 4);
    bf16*  xs_b    = (bf16*) alloc((size_t)BT * DDIM * 2);
    float* x_dbl   = (float*)alloc((size_t)BT * XDBL * 4);
    bf16*  dtp_b   = (bf16*) alloc((size_t)BT * RRANK * 2);
    float* dt_f    = (float*)alloc((size_t)BT * DDIM * 4);
    float* cmat    = (float*)alloc((size_t)BT * DDIM * 4);
    bf16*  y_b     = (bf16*) alloc((size_t)BT * DDIM * 2);
    float* Pbuf    = (float*)alloc((size_t)BATCH * NCHUNK * DDIM * NSTATE * 4);
    float* Qbuf    = (float*)alloc((size_t)BATCH * NCHUNK * DDIM * NSTATE * 4);
    float* Sinit   = (float*)alloc((size_t)BATCH * NCHUNK * DDIM * NSTATE * 4);

    // 1) f32 -> bf16 conversions
    {
        int n;
        n = BT * DDIM;     cvt_bf16_kernel<<<(n+255)/256, 256, 0, stream>>>(x,     xb,      n);
        n = TWOD * DDIM;   cvt_bf16_kernel<<<(n+255)/256, 256, 0, stream>>>(W_in,  W_in_b,  n);
        n = XDBL * DDIM;   cvt_bf16_kernel<<<(n+255)/256, 256, 0, stream>>>(W_x,   W_x_b,   n);
        n = DDIM * RRANK;  cvt_bf16_kernel<<<(n+255)/256, 256, 0, stream>>>(W_dt,  W_dt_b,  n);
        n = DDIM * DDIM;   cvt_bf16_kernel<<<(n+255)/256, 256, 0, stream>>>(W_out, W_out_b, n);
    }

    // 2) xz = x @ W_in^T   (4096 x 2048, K=1024)
    {
        dim3 g((TWOD + 127) / 128, BT / 128);
        wmma_gemm_tn<<<g, 256, 0, stream>>>(xb, W_in_b, xz, BT, TWOD, DDIM, 0, nullptr, 0);
    }

    // 3) depthwise conv + SiLU -> xs (f32 + bf16)
    conv_silu_kernel<<<(BT * DDIM) / 256, 256, 0, stream>>>(xz, conv_w, xs_f, xs_b);

    // 4) x_dbl = xs @ W_x^T  (4096 x 96, K=1024)
    {
        dim3 g((XDBL + 127) / 128, BT / 128);
        wmma_gemm_tn<<<g, 256, 0, stream>>>(xs_b, W_x_b, x_dbl, BT, XDBL, DDIM, 0, nullptr, 0);
    }

    // 5) dt_param -> bf16
    dtp_kernel<<<(BT * RRANK) / 256, 256, 0, stream>>>(x_dbl, dtp_b);

    // 6) dt = softplus(dt_param @ W_dt^T + b_dt) + 1e-5   (4096 x 1024, K=64)
    {
        dim3 g((DDIM + 127) / 128, BT / 128);
        wmma_gemm_tn<<<g, 256, 0, stream>>>(dtp_b, W_dt_b, dt_f, BT, DDIM, RRANK, 1, b_dt, 0);
    }

    // 7) C = Cp @ W_C^T  (K=16, VALU)
    cmat_kernel<<<(BT * DDIM) / 256, 256, 0, stream>>>(x_dbl, W_C, cmat);

    // 8) chunked parallel scan -> y (bf16)
    scan_chunk_kernel<<<(BATCH * NCHUNK * DDIM) / 256, 256, 0, stream>>>(
        x_dbl, dt_f, xs_f, A_log, Pbuf, Qbuf);
    scan_combine_kernel<<<(BATCH * DDIM) / 256, 256, 0, stream>>>(Pbuf, Qbuf, Sinit);
    scan_emit_kernel<<<(BATCH * NCHUNK * DDIM) / 256, 256, 0, stream>>>(
        x_dbl, dt_f, xs_f, cmat, A_log, D_param, Sinit, y_b);

    // 9) out = (y @ W_out^T) * silu(z);  z = xz[..., D:2D]  (aux_ld = 2D)
    {
        dim3 g((DDIM + 127) / 128, BT / 128);
        wmma_gemm_tn<<<g, 256, 0, stream>>>(y_b, W_out_b, out, BT, DDIM, DDIM,
                                            2, xz + DDIM, TWOD);
    }
}
